// EncoderBlock_86380382257156
// MI455X (gfx1250) — compile-verified, matrix operand order checked by reading, not run on hardware
//
#include <hip/hip_runtime.h>
#include <hip/hip_bf16.h>

// Problem constants (match reference)
#define BB 8
#define SS 1024
#define DD 1024
#define HH 16
#define DHH 64
#define FF 4096
#define MMr (BB * SS)   // 8192 rows

typedef __bf16 bf16t;
typedef __attribute__((ext_vector_type(8)))  __bf16 v8bf;
typedef __attribute__((ext_vector_type(16))) __bf16 v16bf;
typedef __attribute__((ext_vector_type(8)))  float  v8f;

union FragBF { v16bf v; v8bf h[2]; };

__device__ inline bf16t f2bf(float f) {
  union { float f; unsigned u; } a; a.f = f;
  unsigned r = a.u + 0x7fffu + ((a.u >> 16) & 1u);   // round-to-nearest-even
  union { unsigned short s; bf16t b; } c; c.s = (unsigned short)(r >> 16);
  return c.b;
}

__device__ inline v8f wmma_bf16(const v16bf& a, const v16bf& b, const v8f& c) {
  // emits v_wmma_f32_16x16x32_bf16
  return __builtin_amdgcn_wmma_f32_16x16x32_bf16(false, a, false, b, (short)0, c,
                                                 false, false);
}

// CDNA5 async global->LDS copy (16B per lane), tracked by ASYNCcnt.
__device__ inline void async_ld_b128(unsigned lds_off, const bf16t* gp) {
  asm volatile("global_load_async_to_lds_b128 %0, %1, off"
               :: "v"(lds_off), "v"((unsigned long long)(uintptr_t)gp)
               : "memory");
}
__device__ inline void wait_async0() {
  asm volatile("s_wait_asynccnt 0x0" ::: "memory");
}
__device__ inline unsigned lds_off_of(const void* p) {
  // generic shared pointer: low 32 bits are the wave-relative LDS address
  return (unsigned)(uintptr_t)p;
}

// ---------------------------------------------------------------- fp32->bf16
__global__ void k_f32_to_bf16(const float* __restrict__ in,
                              bf16t* __restrict__ out, int n) {
  int i = blockIdx.x * blockDim.x + threadIdx.x;
  int stride = gridDim.x * blockDim.x;
  for (; i < n; i += stride) out[i] = f2bf(in[i]);
}

// ------------------------------------------------- fp32 [K][N] -> bf16 [N][K]
__global__ __launch_bounds__(256) void k_transpose_to_bf16(
    const float* __restrict__ in, bf16t* __restrict__ out, int K, int N) {
  __shared__ float tile[64][65];
  const int bx = blockIdx.x * 64;       // n base
  const int by = blockIdx.y * 64;       // k base
  const int tx = threadIdx.x & 63;
  const int ty = threadIdx.x >> 6;      // 0..3
#pragma unroll
  for (int i = 0; i < 16; ++i) {
    int k = ty * 16 + i;
    tile[k][tx] = in[(size_t)(by + k) * N + bx + tx];
  }
  __syncthreads();
#pragma unroll
  for (int i = 0; i < 16; ++i) {
    int n = ty * 16 + i;
    out[(size_t)(bx + n) * K + by + tx] = f2bf(tile[tx][n]);
  }
}

// ---------------------------------------------------------------- GEMM
// C[M,N] = A[M,K](bf16, row-major) @ Wt[N,K](bf16, pre-transposed) + bias
// Double-buffered LDS, K-step 64, async global->LDS staging.
// Epilogue modes: 0 fp32 | 1 bf16 | 2 bf16+ReLU | 3 bf16 [B,H,S,DH] | 4 bf16 [B,H,DH,S]
#define TKB 64           // K elements staged per buffer
#define LDSB 72          // LDS row stride in bf16 (144 B, 16B-aligned)
#define TILE_E (128 * LDSB)

__global__ __launch_bounds__(256) void k_gemm_bf16(
    const bf16t* __restrict__ A, const bf16t* __restrict__ Wt,
    const float* __restrict__ bias, void* __restrict__ outp,
    int Mn, int Nn, int Kn, int mode) {
  __shared__ alignas(16) bf16t sA[2][TILE_E];
  __shared__ alignas(16) bf16t sB[2][TILE_E];
  const int tid  = threadIdx.x;
  const int lane = tid & 31;
  const int wid  = tid >> 5;
  const int wy   = wid >> 1;          // 0..3 -> 32-row strip
  const int wx   = wid & 1;           // 0..1 -> 64-col strip
  const int bm   = blockIdx.y * 128;
  const int bn   = blockIdx.x * 128;
  const int kb   = (lane >> 4) * 8;   // frag K sub-chunk base
  const int l16  = lane & 15;

  const v8f vzero = {};
  v8f acc[2][4];
#pragma unroll
  for (int i = 0; i < 2; ++i)
#pragma unroll
    for (int j = 0; j < 4; ++j) acc[i][j] = vzero;

  // stage one 128x64 A tile + 128x64 B tile into buffer `buf` (async DMA)
  auto stage = [&](int buf, int k0) {
#pragma unroll
    for (int it = 0; it < 4; ++it) {
      int q = tid + it * 256;
      int r = q >> 3, sub = q & 7;          // r: row/col 0..127, sub: 8-elt chunk
      async_ld_b128(lds_off_of(&sA[buf][r * LDSB + sub * 8]),
                    A + (size_t)(bm + r) * Kn + k0 + sub * 8);
      async_ld_b128(lds_off_of(&sB[buf][r * LDSB + sub * 8]),
                    Wt + (size_t)(bn + r) * Kn + k0 + sub * 8);
    }
  };

  stage(0, 0);
  wait_async0();
  __syncthreads();

  const int nk = Kn / TKB;
  for (int kt = 0; kt < nk; ++kt) {
    const int buf = kt & 1;
    if (kt + 1 < nk) stage(buf ^ 1, (kt + 1) * TKB);   // overlap DMA with math

#pragma unroll
    for (int kc = 0; kc < 2; ++kc) {       // two K=32 chunks per staged tile
      FragBF af[2], bfr[4];
#pragma unroll
      for (int mt = 0; mt < 2; ++mt) {
        int r = wy * 32 + mt * 16 + l16;
        const bf16t* p = &sA[buf][r * LDSB + kc * 32 + kb];
        af[mt].h[0] = *(const v8bf*)(p);
        af[mt].h[1] = *(const v8bf*)(p + 16);
      }
#pragma unroll
      for (int nt = 0; nt < 4; ++nt) {
        int c = wx * 64 + nt * 16 + l16;
        const bf16t* p = &sB[buf][c * LDSB + kc * 32 + kb];
        bfr[nt].h[0] = *(const v8bf*)(p);
        bfr[nt].h[1] = *(const v8bf*)(p + 16);
      }
#pragma unroll
      for (int mt = 0; mt < 2; ++mt)
#pragma unroll
        for (int nt = 0; nt < 4; ++nt)
          acc[mt][nt] = wmma_bf16(af[mt].v, bfr[nt].v, acc[mt][nt]);
    }
    wait_async0();
    __syncthreads();
  }

  // epilogue: C layout lane=N, vgpr j -> M = j + 8*(lane>=16)
  float* outF = (float*)outp;
  bf16t* outB = (bf16t*)outp;
  const int rb = (lane & 16) ? 8 : 0;
#pragma unroll
  for (int mt = 0; mt < 2; ++mt) {
#pragma unroll
    for (int nt = 0; nt < 4; ++nt) {
      int col = bn + wx * 64 + nt * 16 + l16;
      float bsv = bias ? bias[col] : 0.f;
#pragma unroll
      for (int j = 0; j < 8; ++j) {
        int row = bm + wy * 32 + mt * 16 + rb + j;
        float v = acc[mt][nt][j] + bsv;
        if (mode == 2) v = v > 0.f ? v : 0.f;
        if (mode == 0) {
          outF[(size_t)row * Nn + col] = v;
        } else if (mode == 3) {       // [B,H,S,DH]
          int b = row >> 10, s = row & 1023, h = col >> 6, d = col & 63;
          outB[(((size_t)(b * HH + h)) * SS + s) * DHH + d] = f2bf(v);
        } else if (mode == 4) {       // [B,H,DH,S]
          int b = row >> 10, s = row & 1023, h = col >> 6, d = col & 63;
          outB[(((size_t)(b * HH + h)) * DHH + d) * SS + s] = f2bf(v);
        } else {
          outB[(size_t)row * Nn + col] = f2bf(v);
        }
      }
    }
  }
}

// ---------------------------------------------------------------- attention
// One wave = one 16-row q tile; online softmax over kv in steps of 32.
__global__ __launch_bounds__(128) void k_attn(
    const bf16t* __restrict__ Q, const bf16t* __restrict__ K,
    const bf16t* __restrict__ Vt, const float* __restrict__ mask,
    bf16t* __restrict__ ctx) {
  __shared__ alignas(16) bf16t sp[4][16 * 40];  // per-wave P buffer 16x32, stride 40
  const int tid  = threadIdx.x;
  const int lane = tid & 31;
  const int w    = tid >> 5;
  const int bh   = blockIdx.x;          // b*H + h
  const int b    = bh >> 4;
  const int h    = bh & 15;
  const int q0   = blockIdx.y * 64 + w * 16;
  const int kb   = (lane >> 4) * 8;
  const int l16  = lane & 15;
  const int rb   = (lane & 16) ? 8 : 0;

  const bf16t* Qp = Q + (size_t)bh * SS * DHH;
  const bf16t* Kp = K + (size_t)bh * SS * DHH;
  const bf16t* Vp = Vt + (size_t)bh * DHH * SS;
  const float* mk = mask + (size_t)b * SS;

  FragBF qf[2];
#pragma unroll
  for (int kc = 0; kc < 2; ++kc) {
    const bf16t* p = Qp + (size_t)(q0 + l16) * DHH + kc * 32 + kb;
    qf[kc].h[0] = *(const v8bf*)(p);
    qf[kc].h[1] = *(const v8bf*)(p + 16);
  }

  const v8f vzero = {};
  float mrow[8], lrow[8];
  v8f acc[4];
#pragma unroll
  for (int j = 0; j < 8; ++j) { mrow[j] = -1e30f; lrow[j] = 0.f; }
#pragma unroll
  for (int nt = 0; nt < 4; ++nt) acc[nt] = vzero;

  const float scale = 0.125f;  // 1/sqrt(64)

  for (int kv = 0; kv < SS; kv += 32) {
    v8f sc[2];
#pragma unroll
    for (int half = 0; half < 2; ++half) {
      sc[half] = vzero;
      int n = kv + half * 16 + l16;
#pragma unroll
      for (int kc = 0; kc < 2; ++kc) {
        FragBF kf;
        const bf16t* p = Kp + (size_t)n * DHH + kc * 32 + kb;
        kf.h[0] = *(const v8bf*)(p);
        kf.h[1] = *(const v8bf*)(p + 16);
        sc[half] = wmma_bf16(qf[kc].v, kf.v, sc[half]);
      }
      float madd = mk[n] * -1e9f;
#pragma unroll
      for (int j = 0; j < 8; ++j) sc[half][j] = sc[half][j] * scale + madd;
    }
    // online softmax: row stats reduced across the 16-lane N group
    float p0[8], p1[8], alpha[8];
#pragma unroll
    for (int j = 0; j < 8; ++j) {
      float t = fmaxf(sc[0][j], sc[1][j]);
      t = fmaxf(t, __shfl_xor(t, 1, 32));
      t = fmaxf(t, __shfl_xor(t, 2, 32));
      t = fmaxf(t, __shfl_xor(t, 4, 32));
      t = fmaxf(t, __shfl_xor(t, 8, 32));
      float mnew = fmaxf(mrow[j], t);
      alpha[j] = __expf(mrow[j] - mnew);
      mrow[j] = mnew;
      p0[j] = __expf(sc[0][j] - mnew);
      p1[j] = __expf(sc[1][j] - mnew);
      float rs = p0[j] + p1[j];
      rs += __shfl_xor(rs, 1, 32);
      rs += __shfl_xor(rs, 2, 32);
      rs += __shfl_xor(rs, 4, 32);
      rs += __shfl_xor(rs, 8, 32);
      lrow[j] = lrow[j] * alpha[j] + rs;
    }
#pragma unroll
    for (int nt = 0; nt < 4; ++nt)
#pragma unroll
      for (int j = 0; j < 8; ++j) acc[nt][j] *= alpha[j];
    // spill P (16x32) to LDS row-major, re-load as WMMA A fragment
#pragma unroll
    for (int j = 0; j < 8; ++j) {
      int r = rb + j;
      sp[w][r * 40 + l16]      = f2bf(p0[j]);
      sp[w][r * 40 + 16 + l16] = f2bf(p1[j]);
    }
    __syncthreads();
    FragBF pf;
    pf.h[0] = *(const v8bf*)(&sp[w][l16 * 40 + kb]);
    pf.h[1] = *(const v8bf*)(&sp[w][l16 * 40 + kb + 16]);
#pragma unroll
    for (int nt = 0; nt < 4; ++nt) {
      FragBF vf;   // Vt[dh][s]: contiguous in kv -> direct b128 frag loads
      const bf16t* p = Vp + (size_t)(nt * 16 + l16) * SS + kv + kb;
      vf.h[0] = *(const v8bf*)(p);
      vf.h[1] = *(const v8bf*)(p + 16);
      acc[nt] = wmma_bf16(pf.v, vf.v, acc[nt]);
    }
    __syncthreads();
  }
  // write merged-head context, bf16
#pragma unroll
  for (int nt = 0; nt < 4; ++nt) {
    int d = nt * 16 + l16;
#pragma unroll
    for (int j = 0; j < 8; ++j) {
      int r = rb + j;
      float v = acc[nt][j] / lrow[j];
      ctx[((size_t)b * SS + q0 + r) * DD + h * DHH + d] = f2bf(v);
    }
  }
}

// ---------------------------------------------------------------- residual+LN
__global__ __launch_bounds__(256) void k_ln(
    const float* __restrict__ a, const float* __restrict__ bres,
    const float* __restrict__ g, const float* __restrict__ beta,
    float* __restrict__ outF, bf16t* __restrict__ outB) {
  __shared__ float red[256];
  const int row = blockIdx.x;
  const int tid = threadIdx.x;
  float v[4];
  float s = 0.f;
#pragma unroll
  for (int i = 0; i < 4; ++i) {
    int c = tid + i * 256;
    v[i] = a[(size_t)row * DD + c] + bres[(size_t)row * DD + c];
    s += v[i];
  }
  red[tid] = s; __syncthreads();
  for (int off = 128; off > 0; off >>= 1) {
    if (tid < off) red[tid] += red[tid + off];
    __syncthreads();
  }
  float mu = red[0] * (1.f / DD);
  __syncthreads();
  s = 0.f;
#pragma unroll
  for (int i = 0; i < 4; ++i) { float d = v[i] - mu; s += d * d; }
  red[tid] = s; __syncthreads();
  for (int off = 128; off > 0; off >>= 1) {
    if (tid < off) red[tid] += red[tid + off];
    __syncthreads();
  }
  float rstd = rsqrtf(red[0] * (1.f / DD) + 1e-6f);
#pragma unroll
  for (int i = 0; i < 4; ++i) {
    int c = tid + i * 256;
    float o = g[c] * (v[i] - mu) * rstd + beta[c];
    if (outF) outF[(size_t)row * DD + c] = o;
    if (outB) outB[(size_t)row * DD + c] = f2bf(o);
  }
}

// ---------------------------------------------------------------- launch
extern "C" void kernel_launch(void* const* d_in, const int* in_sizes, int n_in,
                              void* d_out, int out_size, void* d_ws, size_t ws_size,
                              hipStream_t stream) {
  (void)in_sizes; (void)n_in; (void)out_size; (void)ws_size;
  const float* x    = (const float*)d_in[0];
  const float* mask = (const float*)d_in[1];
  const float* Wq   = (const float*)d_in[2];  const float* bq  = (const float*)d_in[3];
  const float* Wk   = (const float*)d_in[4];  const float* bk  = (const float*)d_in[5];
  const float* Wv   = (const float*)d_in[6];  const float* bv  = (const float*)d_in[7];
  const float* Wo   = (const float*)d_in[8];  const float* bo  = (const float*)d_in[9];
  const float* g1   = (const float*)d_in[10]; const float* be1 = (const float*)d_in[11];
  const float* W1   = (const float*)d_in[12]; const float* bf1 = (const float*)d_in[13];
  const float* W2   = (const float*)d_in[14]; const float* bf2 = (const float*)d_in[15];
  const float* g2   = (const float*)d_in[16]; const float* be2 = (const float*)d_in[17];

  char* wsb = (char*)d_ws;
  size_t off = 0;
  auto alloc = [&](size_t bytes) {
    void* p = wsb + off; off += (bytes + 255) & ~(size_t)255; return p;
  };
  bf16t* xb  = (bf16t*)alloc((size_t)MMr * DD * 2);
  bf16t* wqt = (bf16t*)alloc((size_t)DD * DD * 2);   // all weights transposed [N][K]
  bf16t* wkt = (bf16t*)alloc((size_t)DD * DD * 2);
  bf16t* wvt = (bf16t*)alloc((size_t)DD * DD * 2);
  bf16t* wot = (bf16t*)alloc((size_t)DD * DD * 2);
  bf16t* w1t = (bf16t*)alloc((size_t)FF * DD * 2);   // [F][D]
  bf16t* w2t = (bf16t*)alloc((size_t)DD * FF * 2);   // [D][F]
  bf16t* qh  = (bf16t*)alloc((size_t)MMr * DD * 2);  // [B,H,S,DH]
  bf16t* kh  = (bf16t*)alloc((size_t)MMr * DD * 2);  // [B,H,S,DH]
  bf16t* vt  = (bf16t*)alloc((size_t)MMr * DD * 2);  // [B,H,DH,S]
  bf16t* ctx = (bf16t*)alloc((size_t)MMr * DD * 2);  // merged heads
  float* yf  = (float*)alloc((size_t)MMr * DD * 4);
  float* x1f = (float*)alloc((size_t)MMr * DD * 4);
  bf16t* x1b = (bf16t*)alloc((size_t)MMr * DD * 2);
  bf16t* h1  = qh;   // reuse qh..ctx (64 MB contiguous) for H1 [M,F]
  float* y2f = yf;   // reuse

  dim3 cb(256), cg(4096);
  k_f32_to_bf16<<<cg, cb, 0, stream>>>(x, xb, MMr * DD);
  k_transpose_to_bf16<<<dim3(DD / 64, DD / 64), cb, 0, stream>>>(Wq, wqt, DD, DD);
  k_transpose_to_bf16<<<dim3(DD / 64, DD / 64), cb, 0, stream>>>(Wk, wkt, DD, DD);
  k_transpose_to_bf16<<<dim3(DD / 64, DD / 64), cb, 0, stream>>>(Wv, wvt, DD, DD);
  k_transpose_to_bf16<<<dim3(DD / 64, DD / 64), cb, 0, stream>>>(Wo, wot, DD, DD);
  k_transpose_to_bf16<<<dim3(FF / 64, DD / 64), cb, 0, stream>>>(W1, w1t, DD, FF);
  k_transpose_to_bf16<<<dim3(DD / 64, FF / 64), cb, 0, stream>>>(W2, w2t, FF, DD);

  dim3 gt(256);
  k_gemm_bf16<<<dim3(DD / 128, MMr / 128), gt, 0, stream>>>(xb, wqt, bq, (void*)qh, MMr, DD, DD, 3);
  k_gemm_bf16<<<dim3(DD / 128, MMr / 128), gt, 0, stream>>>(xb, wkt, bk, (void*)kh, MMr, DD, DD, 3);
  k_gemm_bf16<<<dim3(DD / 128, MMr / 128), gt, 0, stream>>>(xb, wvt, bv, (void*)vt, MMr, DD, DD, 4);

  k_attn<<<dim3(BB * HH, SS / 64), dim3(128), 0, stream>>>(qh, kh, vt, mask, ctx);

  k_gemm_bf16<<<dim3(DD / 128, MMr / 128), gt, 0, stream>>>(ctx, wot, bo, (void*)yf, MMr, DD, DD, 0);
  k_ln<<<dim3(MMr), dim3(256), 0, stream>>>(x, yf, g1, be1, x1f, x1b);

  k_gemm_bf16<<<dim3(FF / 128, MMr / 128), gt, 0, stream>>>(x1b, w1t, bf1, (void*)h1, MMr, FF, DD, 2);
  k_gemm_bf16<<<dim3(DD / 128, MMr / 128), gt, 0, stream>>>(h1, w2t, bf2, (void*)y2f, MMr, DD, FF, 0);
  k_ln<<<dim3(MMr), dim3(256), 0, stream>>>(x1f, y2f, g2, be2, (float*)d_out, (bf16t*)nullptr);
}
